// MultiheadAttention_57569741636134
// MI455X (gfx1250) — matrix operand order repaired and checked by reading, not compiled, not run on hardware
//
#include <hip/hip_runtime.h>
#include <stdint.h>

#define B_  2
#define L_  2048
#define D_  1024
#define H_  16
#define DK_ 64
#define SCALE_ 0.35355339059327373f  // 64^-0.25

typedef __attribute__((ext_vector_type(16))) __bf16 v16bf;
typedef __attribute__((ext_vector_type(8)))  float  v8f;

union Frag16 { uint4 u[2]; v16bf v; };

static __device__ inline v8f wmma_bf16(v16bf a, v16bf b, v8f c) {
    // 8 args: (neg_a, A, neg_b, B, c_mod, C, reuse_a, reuse_b)
    return __builtin_amdgcn_wmma_f32_16x16x32_bf16(false, a, false, b, (short)0, c, false, false);
}

// ---------------- f32 -> bf16 conversion ----------------
__global__ __launch_bounds__(256) void cvt_bf16_k(const float* __restrict__ x,
                                                  __bf16* __restrict__ y, int n) {
    int i = blockIdx.x * 256 + threadIdx.x;
    if (i < n) y[i] = (__bf16)x[i];
}

// ---------------- bf16 GEMM: C[M,N] = A[M,K] @ W[N,K]^T ----------------
// M=4096, N=1024, K=1024 fixed. Block = 256 threads (8 waves, 2x4),
// block tile 64x128, wave tile 32x32 (2x2 WMMA frags).
// mode 0: Q epilogue  -> bf16 [B,H,L,DK], *SCALE
// mode 1: K epilogue  -> bf16 [B,H,L,DK], *SCALE
// mode 2: V epilogue  -> bf16 [B,H,DK,L] (transposed)
// mode 3: plain f32 row-major [M,N] (final output proj)
__global__ __launch_bounds__(256) void gemm_bf16_k(const __bf16* __restrict__ A,
                                                   const __bf16* __restrict__ W,
                                                   void* __restrict__ out, int mode) {
    const int K = 1024;
    int tid = threadIdx.x, wid = tid >> 5, lane = tid & 31;
    int ln = lane & 15, hi = lane >> 4;
    int wm = wid >> 2, wn = wid & 3;
    int rb = blockIdx.x * 64 + wm * 32;   // row base
    int cb = blockIdx.y * 128 + wn * 32;  // col base

    const v8f zero = {0.f,0.f,0.f,0.f,0.f,0.f,0.f,0.f};
    v8f acc[2][2] = {{zero, zero}, {zero, zero}};

    int klo = hi ? 16 : 0;   // A-frag K-half offset
    int kco = hi ? 16 : 0;   // B-frag K-half offset

    for (int k0 = 0; k0 < K; k0 += 32) {
        Frag16 af[2], bf[2];
#pragma unroll
        for (int i = 0; i < 2; ++i) {
            const __bf16* p = A + (size_t)(rb + i * 16 + ln) * K + k0 + klo;
            af[i].u[0] = *(const uint4*)p;
            af[i].u[1] = *(const uint4*)(p + 8);
        }
#pragma unroll
        for (int j = 0; j < 2; ++j) {
            const __bf16* p = W + (size_t)(cb + j * 16 + ln) * K + k0 + kco;
            bf[j].u[0] = *(const uint4*)p;
            bf[j].u[1] = *(const uint4*)(p + 8);
        }
        if (k0 + 32 < K) { // hint next tiles into cache (global_prefetch_b8)
            __builtin_prefetch(A + (size_t)(rb + ln) * K + k0 + 32, 0, 1);
            __builtin_prefetch(W + (size_t)(cb + ln) * K + k0 + 32, 0, 1);
        }
#pragma unroll
        for (int i = 0; i < 2; ++i)
#pragma unroll
            for (int j = 0; j < 2; ++j)
                acc[i][j] = wmma_bf16(af[i].v, bf[j].v, acc[i][j]);
    }

#pragma unroll
    for (int i = 0; i < 2; ++i)
#pragma unroll
        for (int j = 0; j < 2; ++j) {
            int n = cb + j * 16 + ln;
            int h = n >> 6, dk = n & 63;
#pragma unroll
            for (int r = 0; r < 8; ++r) {
                int m = rb + i * 16 + r + hi * 8;
                float val = acc[i][j][r];
                if (mode == 3) {
                    ((float*)out)[(size_t)m * D_ + n] = val;
                } else {
                    int b = m >> 11, l = m & (L_ - 1);
                    if (mode == 2) {
                        ((__bf16*)out)[(((size_t)(b * H_ + h)) * DK_ + dk) * L_ + l] = (__bf16)val;
                    } else {
                        ((__bf16*)out)[(((size_t)(b * H_ + h)) * L_ + l) * DK_ + dk] =
                            (__bf16)(val * SCALE_);
                    }
                }
            }
        }
}

// ---------------- fused flash attention ----------------
// Grid: B*H*(L/64) blocks, 128 threads (4 waves). Each wave: 16 query rows,
// streams key blocks of 32, online softmax, O accumulated in WMMA f32 frags.
__global__ __launch_bounds__(128) void flash_attn_k(const __bf16* __restrict__ Qp,
                                                    const __bf16* __restrict__ Kp,
                                                    const __bf16* __restrict__ VTp,
                                                    const float* __restrict__ bias,
                                                    __bf16* __restrict__ AO) {
    __shared__ __bf16 plds[4][16][32];  // per-wave P staging (C-layout -> A-layout)

    int tid = threadIdx.x, wid = tid >> 5, lane = tid & 31;
    int ln = lane & 15, hi = lane >> 4;
    int qt = blockIdx.x & 31;
    int h  = (blockIdx.x >> 5) & 15;
    int b  = blockIdx.x >> 9;
    int q0 = qt * 64 + wid * 16;

    const __bf16* Qb = Qp  + ((size_t)(b * H_ + h)) * L_ * DK_;
    const __bf16* Kb = Kp  + ((size_t)(b * H_ + h)) * L_ * DK_;
    const __bf16* Vb = VTp + ((size_t)(b * H_ + h)) * DK_ * L_;
    const float*  Pb = bias + ((size_t)(b * H_ + h)) * L_ * L_;

    int klo = hi ? 16 : 0;  // A-frag half
    int kco = hi ? 16 : 0;  // B-frag half

    // Q A-fragments for dk 0..31 and 32..63 (rows q0..q0+15)
    Frag16 qa0, qa1;
    {
        const __bf16* p = Qb + (size_t)(q0 + ln) * DK_ + klo;
        qa0.u[0] = *(const uint4*)p;        qa0.u[1] = *(const uint4*)(p + 8);
        qa1.u[0] = *(const uint4*)(p + 32); qa1.u[1] = *(const uint4*)(p + 40);
    }

    const v8f zero = {0.f,0.f,0.f,0.f,0.f,0.f,0.f,0.f};
    v8f o[4] = {zero, zero, zero, zero};
    float mrow[8], lrow[8];
#pragma unroll
    for (int j = 0; j < 8; ++j) { mrow[j] = -3.0e38f; lrow[j] = 0.f; }

    for (int j0 = 0; j0 < L_; j0 += 32) {
        // K B-fragments: keys j0..+15 and j0+16..+31, dk halves 0..31 / 32..63
        Frag16 k00, k01, k10, k11;
        {
            const __bf16* p = Kb + (size_t)(j0 + ln) * DK_ + kco;
            k00.u[0] = *(const uint4*)p;        k00.u[1] = *(const uint4*)(p + 8);
            k01.u[0] = *(const uint4*)(p + 32); k01.u[1] = *(const uint4*)(p + 40);
        }
        {
            const __bf16* p = Kb + (size_t)(j0 + 16 + ln) * DK_ + kco;
            k10.u[0] = *(const uint4*)p;        k10.u[1] = *(const uint4*)(p + 8);
            k11.u[0] = *(const uint4*)(p + 32); k11.u[1] = *(const uint4*)(p + 40);
        }
        v8f s0 = zero, s1 = zero;
        s0 = wmma_bf16(qa0.v, k00.v, s0);
        s0 = wmma_bf16(qa1.v, k01.v, s0);
        s1 = wmma_bf16(qa0.v, k10.v, s1);
        s1 = wmma_bf16(qa1.v, k11.v, s1);

        // + posbias (the single 537 MB stream)
#pragma unroll
        for (int j = 0; j < 8; ++j) {
            const float* bp = Pb + (size_t)(q0 + j + hi * 8) * L_ + j0 + ln;
            s0[j] += bp[0];
            s1[j] += bp[16];
        }

        // online softmax per row (rows live across the 16-lane half)
#pragma unroll
        for (int j = 0; j < 8; ++j) {
            float t = fmaxf(s0[j], s1[j]);
            t = fmaxf(t, __shfl_xor(t, 1, 32));
            t = fmaxf(t, __shfl_xor(t, 2, 32));
            t = fmaxf(t, __shfl_xor(t, 4, 32));
            t = fmaxf(t, __shfl_xor(t, 8, 32));
            float mn = fmaxf(mrow[j], t);
            float alpha = __expf(mrow[j] - mn);
            float p0v = __expf(s0[j] - mn);
            float p1v = __expf(s1[j] - mn);
            float rs = p0v + p1v;
            rs += __shfl_xor(rs, 1, 32);
            rs += __shfl_xor(rs, 2, 32);
            rs += __shfl_xor(rs, 4, 32);
            rs += __shfl_xor(rs, 8, 32);
            lrow[j] = lrow[j] * alpha + rs;
            mrow[j] = mn;
#pragma unroll
            for (int t2 = 0; t2 < 4; ++t2) o[t2][j] *= alpha;
            plds[wid][j + hi * 8][ln]      = (__bf16)p0v;
            plds[wid][j + hi * 8][16 + ln] = (__bf16)p1v;
        }

        asm volatile("s_wait_dscnt 0" ::: "memory");

        // re-read P as an A-fragment (16x32 over keys)
        Frag16 pa;
        {
            const __bf16* p = &plds[wid][ln][klo];
            pa.u[0] = *(const uint4*)p;
            pa.u[1] = *(const uint4*)(p + 8);
        }

        // O += P @ V  (V^T layout -> contiguous B-frag loads over keys)
#pragma unroll
        for (int t2 = 0; t2 < 4; ++t2) {
            Frag16 vbf;
            const __bf16* p = Vb + (size_t)(t2 * 16 + ln) * L_ + j0 + kco;
            vbf.u[0] = *(const uint4*)p;
            vbf.u[1] = *(const uint4*)(p + 8);
            o[t2] = wmma_bf16(pa.v, vbf.v, o[t2]);
        }
    }

    // normalize and store attn output as bf16 [B, L, H*DK]
#pragma unroll
    for (int j = 0; j < 8; ++j) {
        float inv = 1.0f / lrow[j];
        int rowg = q0 + j + hi * 8;
        __bf16* dst = AO + ((size_t)(b * L_ + rowg)) * D_ + h * DK_;
#pragma unroll
        for (int t2 = 0; t2 < 4; ++t2)
            dst[t2 * 16 + ln] = (__bf16)(o[t2][j] * inv);
    }
}

// ---------------- host launch ----------------
extern "C" void kernel_launch(void* const* d_in, const int* in_sizes, int n_in,
                              void* d_out, int out_size, void* d_ws, size_t ws_size,
                              hipStream_t stream) {
    const float* query = (const float*)d_in[0];
    const float* key   = (const float*)d_in[1];
    const float* value = (const float*)d_in[2];
    const float* pbias = (const float*)d_in[3];
    const float* Wq    = (const float*)d_in[4];
    const float* Wk    = (const float*)d_in[5];
    const float* Wv    = (const float*)d_in[6];
    const float* Wo    = (const float*)d_in[7];

    const int NBLD = B_ * L_ * D_;   // 4194304
    const int NDD  = D_ * D_;        // 1048576

    char* ws = (char*)d_ws;
    __bf16* xq  = (__bf16*)ws;                 ws += (size_t)NBLD * 2;
    __bf16* xk  = (__bf16*)ws;                 ws += (size_t)NBLD * 2;
    __bf16* xv  = (__bf16*)ws;                 ws += (size_t)NBLD * 2;
    __bf16* wqb = (__bf16*)ws;                 ws += (size_t)NDD * 2;
    __bf16* wkb = (__bf16*)ws;                 ws += (size_t)NDD * 2;
    __bf16* wvb = (__bf16*)ws;                 ws += (size_t)NDD * 2;
    __bf16* wob = (__bf16*)ws;                 ws += (size_t)NDD * 2;
    __bf16* Qb  = (__bf16*)ws;                 ws += (size_t)NBLD * 2;  // [B,H,L,DK]
    __bf16* Kb  = (__bf16*)ws;                 ws += (size_t)NBLD * 2;  // [B,H,L,DK]
    __bf16* VTb = (__bf16*)ws;                 ws += (size_t)NBLD * 2;  // [B,H,DK,L]
    __bf16* AOb = (__bf16*)ws;                 ws += (size_t)NBLD * 2;  // [B,L,D]

    // f32 -> bf16 conversions
    cvt_bf16_k<<<NBLD / 256, 256, 0, stream>>>(query, xq, NBLD);
    cvt_bf16_k<<<NBLD / 256, 256, 0, stream>>>(key,   xk, NBLD);
    cvt_bf16_k<<<NBLD / 256, 256, 0, stream>>>(value, xv, NBLD);
    cvt_bf16_k<<<NDD  / 256, 256, 0, stream>>>(Wq, wqb, NDD);
    cvt_bf16_k<<<NDD  / 256, 256, 0, stream>>>(Wk, wkb, NDD);
    cvt_bf16_k<<<NDD  / 256, 256, 0, stream>>>(Wv, wvb, NDD);
    cvt_bf16_k<<<NDD  / 256, 256, 0, stream>>>(Wo, wob, NDD);

    dim3 ggrid(64, 8, 1);  // M/64 x N/128
    gemm_bf16_k<<<ggrid, 256, 0, stream>>>(xq, wqb, Qb,  0);  // Q proj (+scale)
    gemm_bf16_k<<<ggrid, 256, 0, stream>>>(xk, wkb, Kb,  1);  // K proj (+scale)
    gemm_bf16_k<<<ggrid, 256, 0, stream>>>(xv, wvb, VTb, 2);  // V proj (transposed)

    flash_attn_k<<<B_ * H_ * (L_ / 64), 128, 0, stream>>>(Qb, Kb, VTb, pbias, AOb);

    gemm_bf16_k<<<ggrid, 256, 0, stream>>>(AOb, wob, d_out, 3);  // output proj -> f32
}
